// KANLinear_22522808500876
// MI455X (gfx1250) — matrix operand order repaired and checked by reading, not compiled, not run on hardware
//
#include <hip/hip_runtime.h>
#include <hip/hip_bf16.h>

// ---------------- problem constants ----------------
#define BATCH   8192
#define IN_F    1024
#define OUT_F   1024
#define NBASIS  8           // GRID_SIZE + SPLINE_ORDER = 5 + 3
#define NKNOT   12          // GRID_SIZE + 2*SPLINE_ORDER + 1
#define KTOT    (IN_F + IN_F * NBASIS)   // 9216

typedef __attribute__((ext_vector_type(16))) __bf16 v16bf;
typedef __attribute__((ext_vector_type(8)))  float  v8f;
typedef __attribute__((ext_vector_type(4)))  int    v4i;

union frag16 { v16bf v; uint4 u[2]; };

#if defined(__gfx1250__) && __has_builtin(__builtin_amdgcn_global_load_async_to_lds_b128)
#define USE_ASYNC 1
#else
#define USE_ASYNC 0
#endif

// cast helpers for the async builtin's exact pointer types
#define GAS1(p) ((__attribute__((address_space(1))) v4i*)(void*)(p))
#define LAS3(p) ((__attribute__((address_space(3))) v4i*)(p))

__device__ __forceinline__ void async_wait0() {
#if defined(__gfx1250__)
#if __has_builtin(__builtin_amdgcn_s_wait_asynccnt)
    __builtin_amdgcn_s_wait_asynccnt(0);
#else
    asm volatile("s_wait_asynccnt 0x0" ::: "memory");
#endif
#endif
}

// order-preserving float <-> uint mapping (min/max via integer atomics)
__device__ __forceinline__ unsigned f2key(float f) {
    unsigned b = __float_as_uint(f);
    return (b & 0x80000000u) ? ~b : (b | 0x80000000u);
}
__device__ __forceinline__ float key2f(unsigned k) {
    unsigned b = (k & 0x80000000u) ? (k & 0x7FFFFFFFu) : ~k;
    return __uint_as_float(b);
}

// ---------------- kernel 0: init min/max keys ----------------
__global__ void kan_init_minmax(unsigned* keys) {
    keys[0] = 0xFFFFFFFFu;   // min key
    keys[1] = 0x00000000u;   // max key
}

// ---------------- kernel 1: global min/max of x ----------------
__global__ void kan_minmax(const float* __restrict__ x, unsigned* keys, int n) {
    __shared__ float smn[256], smx[256];
    int tid = threadIdx.x;
    float mn = INFINITY, mx = -INFINITY;
    for (int i = blockIdx.x * blockDim.x + tid; i < n; i += gridDim.x * blockDim.x) {
        float v = x[i];
        mn = fminf(mn, v);
        mx = fmaxf(mx, v);
    }
    smn[tid] = mn; smx[tid] = mx;
    __syncthreads();
    for (int s = 128; s > 0; s >>= 1) {
        if (tid < s) {
            smn[tid] = fminf(smn[tid], smn[tid + s]);
            smx[tid] = fmaxf(smx[tid], smx[tid + s]);
        }
        __syncthreads();
    }
    if (tid == 0) {
        atomicMin(&keys[0], f2key(smn[0]));
        atomicMax(&keys[1], f2key(smx[0]));
    }
}

// ---------------- kernel 2: pack A = [silu(x) | basis(x_norm)] in bf16 ----------------
__global__ void kan_pack_A(const float* __restrict__ x,
                           const float* __restrict__ grid,
                           const unsigned* __restrict__ keys,
                           __bf16* __restrict__ A) {
    int idx = blockIdx.x * blockDim.x + threadIdx.x;   // over BATCH*IN_F
    int b = idx >> 10;          // / IN_F
    int i = idx & 1023;         // % IN_F
    float xv = x[idx];

    // silu part
    float s = xv * (1.0f / (1.0f + __expf(-xv)));
    A[(long)b * KTOT + i] = (__bf16)s;

    // normalize
    float xmin = key2f(keys[0]);
    float xmax = key2f(keys[1]);
    float xn = (xv - xmin) / (xmax - xmin + 1e-8f) * 2.0f - 1.0f;

    // knots for feature i
    float g[NKNOT];
    #pragma unroll
    for (int t = 0; t < NKNOT; ++t) g[t] = grid[i * NKNOT + t];

    // Cox-de-Boor, order 3
    float Bv[NKNOT - 1];
    #pragma unroll
    for (int t = 0; t < NKNOT - 1; ++t)
        Bv[t] = (xn >= g[t] && xn < g[t + 1]) ? 1.0f : 0.0f;
    #pragma unroll
    for (int k = 1; k <= 3; ++k) {
        #pragma unroll
        for (int t = 0; t < NKNOT - 1 - k; ++t) {
            float left  = (xn - g[t])         / (g[t + k]     - g[t]);
            float right = (g[t + k + 1] - xn) / (g[t + k + 1] - g[t + 1]);
            Bv[t] = left * Bv[t] + right * Bv[t + 1];
        }
    }

    // 8 bf16 basis values -> one 16-byte store
    union { __bf16 h[8]; uint4 u; } out;
    #pragma unroll
    for (int j = 0; j < NBASIS; ++j) out.h[j] = (__bf16)Bv[j];
    *(uint4*)(A + (long)b * KTOT + IN_F + i * NBASIS) = out.u;
}

// ---------------- kernel 3: pack W = [base_weight | spline_weight*scaler] in bf16 ----------------
__global__ void kan_pack_W(const float* __restrict__ bw,
                           const float* __restrict__ sw,
                           const float* __restrict__ sc,
                           __bf16* __restrict__ W) {
    int idx = blockIdx.x * blockDim.x + threadIdx.x;   // over OUT_F*IN_F
    int o = idx >> 10;
    int i = idx & 1023;

    W[(long)o * KTOT + i] = (__bf16)bw[idx];

    float scale = sc[idx];
    union { __bf16 h[8]; uint4 u; } out;
    #pragma unroll
    for (int j = 0; j < NBASIS; ++j)
        out.h[j] = (__bf16)(sw[(long)idx * NBASIS + j] * scale);
    *(uint4*)(W + (long)o * KTOT + IN_F + i * NBASIS) = out.u;
}

// ---------------- kernel 4: bf16 WMMA GEMM: C(8192x1024) = A(8192xK) * W(1024xK)^T ----------------
#define LDT 40   // LDS tile row stride in bf16 (32 + 8 pad -> conflict-free 16B reads)

__global__ __launch_bounds__(256, 1)
void kan_gemm(const __bf16* __restrict__ A,
              const __bf16* __restrict__ W,
              float* __restrict__ C) {
#if USE_ASYNC
    __shared__ __bf16 Asm[2][128 * LDT];
    __shared__ __bf16 Bsm[2][128 * LDT];
#else
    __shared__ __bf16 Asm[1][128 * LDT];
    __shared__ __bf16 Bsm[1][128 * LDT];
#endif

    const int tid  = threadIdx.x;
    const int lane = tid & 31;
    const int wave = tid >> 5;        // 0..7
    const int wm   = wave & 1;        // wave M position (0..1), tile 64 rows
    const int wn   = wave >> 1;       // wave N position (0..3), tile 32 cols
    const int half = lane >> 4;       // lane group
    const int lrow = lane & 15;

    const int blockN = blockIdx.x;    // 0..7   (N-major for L2 reuse of A rows)
    const int blockM = blockIdx.y;    // 0..63
    const long aBase = (long)(blockM * 128) * KTOT;
    const long bBase = (long)(blockN * 128) * KTOT;

    // this thread's two 16B chunks of each 128x32 tile
    const int chunk0 = tid * 2;
    const int r0  = chunk0 >> 2,       c80 = (chunk0 & 3) * 8;
    const int r1  = (chunk0 + 1) >> 2, c81 = ((chunk0 + 1) & 3) * 8;

    v8f acc[4][2];
    #pragma unroll
    for (int i = 0; i < 4; ++i)
        #pragma unroll
        for (int j = 0; j < 2; ++j)
            acc[i][j] = (v8f){0.f, 0.f, 0.f, 0.f, 0.f, 0.f, 0.f, 0.f};

#if USE_ASYNC
    // issue async copy of one (A,B) tile pair into LDS buffer `buf`
    #define ISSUE_COPY(buf, kb)                                                        \
        do {                                                                           \
            __builtin_amdgcn_global_load_async_to_lds_b128(                            \
                GAS1(A + aBase + (long)r0 * KTOT + (kb) + c80),                        \
                LAS3(&Asm[buf][r0 * LDT + c80]), 0, 0);                                \
            __builtin_amdgcn_global_load_async_to_lds_b128(                            \
                GAS1(A + aBase + (long)r1 * KTOT + (kb) + c81),                        \
                LAS3(&Asm[buf][r1 * LDT + c81]), 0, 0);                                \
            __builtin_amdgcn_global_load_async_to_lds_b128(                            \
                GAS1(W + bBase + (long)r0 * KTOT + (kb) + c80),                        \
                LAS3(&Bsm[buf][r0 * LDT + c80]), 0, 0);                                \
            __builtin_amdgcn_global_load_async_to_lds_b128(                            \
                GAS1(W + bBase + (long)r1 * KTOT + (kb) + c81),                        \
                LAS3(&Bsm[buf][r1 * LDT + c81]), 0, 0);                                \
        } while (0)

    ISSUE_COPY(0, 0);
    int cur = 0;
#endif

    for (int kb = 0; kb < KTOT; kb += 32) {
#if USE_ASYNC
        async_wait0();          // this wave's copies into `cur` are done
        __syncthreads();        // all waves' copies visible; prev-buffer readers done
        if (kb + 32 < KTOT) {
            if (cur == 0) ISSUE_COPY(1, kb + 32); else ISSUE_COPY(0, kb + 32);
        }
        const __bf16* Abuf = Asm[cur];
        const __bf16* Bbuf = Bsm[cur];
#else
        // sync staging path: global -> VGPR -> LDS
        {
            uint4 av0 = *(const uint4*)(A + aBase + (long)r0 * KTOT + kb + c80);
            uint4 av1 = *(const uint4*)(A + aBase + (long)r1 * KTOT + kb + c81);
            uint4 bv0 = *(const uint4*)(W + bBase + (long)r0 * KTOT + kb + c80);
            uint4 bv1 = *(const uint4*)(W + bBase + (long)r1 * KTOT + kb + c81);
            *(uint4*)(&Asm[0][r0 * LDT + c80]) = av0;
            *(uint4*)(&Asm[0][r1 * LDT + c81]) = av1;
            *(uint4*)(&Bsm[0][r0 * LDT + c80]) = bv0;
            *(uint4*)(&Bsm[0][r1 * LDT + c81]) = bv1;
        }
        __syncthreads();
        const __bf16* Abuf = Asm[0];
        const __bf16* Bbuf = Bsm[0];
#endif

        // A fragments: lane holds row m; elems j=0..7 -> k=half*8+j, j=8..15 -> k=16+half*8+(j-8)
        v16bf afrag[4];
        #pragma unroll
        for (int i = 0; i < 4; ++i) {
            const __bf16* p = Abuf + (wm * 64 + i * 16 + lrow) * LDT + half * 8;
            frag16 t;
            t.u[0] = *(const uint4*)(p);
            t.u[1] = *(const uint4*)(p + 16);
            afrag[i] = t.v;
        }
        // B fragments: lane holds col n; elems j=0..7 -> k=half*8+j, j=8..15 -> k=16+half*8+(j-8)
        v16bf bfrag[2];
        #pragma unroll
        for (int j = 0; j < 2; ++j) {
            const __bf16* p = Bbuf + (wn * 32 + j * 16 + lrow) * LDT + half * 8;
            frag16 t;
            t.u[0] = *(const uint4*)(p);
            t.u[1] = *(const uint4*)(p + 16);
            bfrag[j] = t.v;
        }

        #pragma unroll
        for (int i = 0; i < 4; ++i)
            #pragma unroll
            for (int j = 0; j < 2; ++j)
                acc[i][j] = __builtin_amdgcn_wmma_f32_16x16x32_bf16(
                    false, afrag[i], false, bfrag[j],
                    (short)0, acc[i][j], false, false);

#if USE_ASYNC
        cur ^= 1;
#else
        __syncthreads();
#endif
    }

    // C/D layout: vgpr r, lane half -> row = r + half*8; lane lrow -> col
    #pragma unroll
    for (int i = 0; i < 4; ++i) {
        #pragma unroll
        for (int j = 0; j < 2; ++j) {
            int row0 = blockM * 128 + wm * 64 + i * 16 + half * 8;
            int col  = blockN * 128 + wn * 32 + j * 16 + lrow;
            #pragma unroll
            for (int r = 0; r < 8; ++r)
                C[(long)(row0 + r) * OUT_F + col] = acc[i][j][r];
        }
    }
}

// ---------------- launcher ----------------
extern "C" void kernel_launch(void* const* d_in, const int* in_sizes, int n_in,
                              void* d_out, int out_size, void* d_ws, size_t ws_size,
                              hipStream_t stream) {
    const float* x    = (const float*)d_in[0];   // (8192,1024)
    const float* grid = (const float*)d_in[1];   // (1024,12)
    const float* bw   = (const float*)d_in[2];   // (1024,1024)
    const float* sw   = (const float*)d_in[3];   // (1024,1024,8)
    const float* sc   = (const float*)d_in[4];   // (1024,1024)
    float* out = (float*)d_out;                  // (8192,1024)

    // workspace layout
    unsigned* keys = (unsigned*)d_ws;
    __bf16* A = (__bf16*)((char*)d_ws + 256);
    __bf16* W = A + (long)BATCH * KTOT;          // A: 151 MB, W: 18.9 MB (bf16)

    kan_init_minmax<<<1, 1, 0, stream>>>(keys);
    kan_minmax<<<1024, 256, 0, stream>>>(x, keys, BATCH * IN_F);
    kan_pack_A<<<(BATCH * IN_F) / 256, 256, 0, stream>>>(x, grid, keys, A);
    kan_pack_W<<<(OUT_F * IN_F) / 256, 256, 0, stream>>>(bw, sw, sc, W);

    dim3 g(OUT_F / 128, BATCH / 128);            // (8, 64): N-major scheduling
    kan_gemm<<<g, 256, 0, stream>>>(A, W, out);
}
